// Attention_6124623364158
// MI455X (gfx1250) — compile-verified, matrix operand order checked by reading, not run on hardware
//
#include <hip/hip_runtime.h>

// ---------------------------------------------------------------------------
// Attention block for MI455X (gfx1250, wave32, WMMA bf16 -> f32 accumulate)
//   B=4, N=2048, DIM=768, H=12, hd=64
// Pipeline: cvt->bf16, QKV GEMM, flash attention (TDM-fed K tiles), proj GEMM
// ---------------------------------------------------------------------------

#define BATCH   4
#define SEQ     2048
#define DIM     768
#define NHEAD   12
#define HDIM    64
#define QKVDIM  (3 * DIM)          // 2304
#define MROWS   (BATCH * SEQ)      // 8192
#define ATTN_SCALE 0.125f          // 64^-0.5

typedef __attribute__((ext_vector_type(16))) __bf16          v16bf;
typedef __attribute__((ext_vector_type(8)))  float           v8f;
typedef __attribute__((ext_vector_type(8)))  unsigned short  v8us;
typedef __attribute__((ext_vector_type(4)))  unsigned int    v4u;
typedef __attribute__((ext_vector_type(4)))  int             v4i;
typedef __attribute__((ext_vector_type(8)))  int             v8i_t;

#if defined(__has_builtin)
#  if __has_builtin(__builtin_amdgcn_tensor_load_to_lds) && \
      __has_builtin(__builtin_amdgcn_s_wait_tensorcnt)
#    define HAVE_TDM 1
#  endif
#endif
#ifndef HAVE_TDM
#  define HAVE_TDM 0
#endif

union Frag { v16bf bf; v8us us[2]; };

// fp32 -> bf16 (round to nearest even), raw bits in ushort storage
__device__ __forceinline__ unsigned short bf16_of(float f) {
    unsigned int u = __float_as_uint(f);
    u += 0x7FFFu + ((u >> 16) & 1u);
    return (unsigned short)(u >> 16);
}

// Load one 16x32 bf16 A/B fragment row for this lane.
// lo = 8*(lane>>4): lanes 0-15 hold K {0..7,16..23}, lanes 16-31 {8..15,24..31}
__device__ __forceinline__ v16bf load_row_frag(const unsigned short* rowPtr, int lo) {
    Frag f;
    f.us[0] = *reinterpret_cast<const v8us*>(rowPtr + lo);
    f.us[1] = *reinterpret_cast<const v8us*>(rowPtr + 16 + lo);
    return f.bf;
}

__device__ __forceinline__ v8f wmma_bf16(v16bf a, v16bf b, v8f c) {
    return __builtin_amdgcn_wmma_f32_16x16x32_bf16(
        false, a, false, b, (short)0, c, false, false);
}

__device__ __forceinline__ float rowmax16(float v) {
    v = fmaxf(v, __shfl_xor(v, 1, 32));
    v = fmaxf(v, __shfl_xor(v, 2, 32));
    v = fmaxf(v, __shfl_xor(v, 4, 32));
    v = fmaxf(v, __shfl_xor(v, 8, 32));
    return v;
}
__device__ __forceinline__ float rowsum16(float v) {
    v += __shfl_xor(v, 1, 32);
    v += __shfl_xor(v, 2, 32);
    v += __shfl_xor(v, 4, 32);
    v += __shfl_xor(v, 8, 32);
    return v;
}

#if HAVE_TDM
// Issue one TDM 2D tile load: rows x cols bf16 elements, row stride in
// elements, packed contiguously into LDS at ldst. D# built per ISA ch.8.
__device__ __forceinline__ void tdm_load_tile_2d(
    const unsigned short* gsrc, unsigned short* ldst,
    unsigned rows, unsigned cols, unsigned stride_elems, unsigned rows_total)
{
    unsigned long long ga = (unsigned long long)(size_t)gsrc;
    unsigned lds_off = (unsigned)(size_t)ldst;       // addr[31:0] == LDS offset
    v4u g0;
    g0.x = 1u;                                        // count=1 (valid), user mode
    g0.y = lds_off;                                   // lds_addr
    g0.z = (unsigned)(ga & 0xFFFFFFFFull);            // global_addr[31:0]
    g0.w = (unsigned)((ga >> 32) & 0x1FFFFFFull) | (2u << 30);  // addr[56:32] | type=2
    v8i_t g1;
    g1[0] = (int)(1u << 16);                          // data_size=1 -> 2 bytes
    g1[1] = (int)(cols << 16);                        // tensor_dim0 lo16
    g1[2] = (int)(((cols >> 16) & 0xFFFFu) | ((rows_total & 0xFFFFu) << 16)); // dim0 hi | dim1 lo
    g1[3] = (int)(((rows_total >> 16) & 0xFFFFu) | (cols << 16));             // dim1 hi | tile_dim0
    g1[4] = (int)(rows & 0xFFFFu);                    // tile_dim1 (tile_dim2=0)
    g1[5] = (int)stride_elems;                        // tensor_dim0_stride lo32
    g1[6] = 0;                                        // stride hi | dim1_stride lo
    g1[7] = 0;
    v4i g2; g2.x = 1; g2.y = 1; g2.z = 0; g2.w = 0;   // tensor_dim2=1, dim3=1
    v4i g3; g3.x = 0; g3.y = (int)(1u << 16); g3.z = 0; g3.w = 0; // tensor_dim4=1
#if __clang_major__ >= 23
    v8i_t gx; gx[0]=0; gx[1]=0; gx[2]=0; gx[3]=0; gx[4]=0; gx[5]=0; gx[6]=0; gx[7]=0;
    __builtin_amdgcn_tensor_load_to_lds(g0, g1, g2, g3, gx, 0);
#else
    __builtin_amdgcn_tensor_load_to_lds(g0, g1, g2, g3, 0);
#endif
}
#endif

// ---------------------------------------------------------------------------
// Kernel 0: fp32 -> bf16 conversion
// ---------------------------------------------------------------------------
__global__ void cvt_f32_bf16(const float* __restrict__ src,
                             unsigned short* __restrict__ dst, int n) {
    int i = blockIdx.x * blockDim.x + threadIdx.x;
    if (i < n) dst[i] = bf16_of(src[i]);
}

// ---------------------------------------------------------------------------
// GEMM core: 64 (M) x 32 (N) wave tile, K stepped by 64 per iteration with
// two structurally distinct fragment buffer sets (no rotation copies, no
// WMMA->mov WAR hazard nops). Loads for the next half-step are issued before
// the WMMAs of the current one.
// ---------------------------------------------------------------------------
__device__ __forceinline__ void gemm_64x32_body(
    const unsigned short* ar0, const unsigned short* ar1,
    const unsigned short* ar2, const unsigned short* ar3,
    const unsigned short* br0, const unsigned short* br1,
    int lo, v8f acc[2][4])
{
    v16bf aA[4], bA[2], aB[4], bB[2];
    aA[0] = load_row_frag(ar0, lo);
    aA[1] = load_row_frag(ar1, lo);
    aA[2] = load_row_frag(ar2, lo);
    aA[3] = load_row_frag(ar3, lo);
    bA[0] = load_row_frag(br0, lo);
    bA[1] = load_row_frag(br1, lo);

    for (int kk = 0; kk < DIM; kk += 64) {
        // load odd half-step (kk+32) -- always in range (DIM % 64 == 0)
        const int k1 = kk + 32;
        aB[0] = load_row_frag(ar0 + k1, lo);
        aB[1] = load_row_frag(ar1 + k1, lo);
        aB[2] = load_row_frag(ar2 + k1, lo);
        aB[3] = load_row_frag(ar3 + k1, lo);
        bB[0] = load_row_frag(br0 + k1, lo);
        bB[1] = load_row_frag(br1 + k1, lo);

        #pragma unroll
        for (int i = 0; i < 4; ++i) acc[0][i] = wmma_bf16(aA[i], bA[0], acc[0][i]);
        #pragma unroll
        for (int i = 0; i < 4; ++i) acc[1][i] = wmma_bf16(aA[i], bA[1], acc[1][i]);

        // load next even half-step (kk+64), guarded on the final iteration
        const int k2 = kk + 64;
        if (k2 < DIM) {
            aA[0] = load_row_frag(ar0 + k2, lo);
            aA[1] = load_row_frag(ar1 + k2, lo);
            aA[2] = load_row_frag(ar2 + k2, lo);
            aA[3] = load_row_frag(ar3 + k2, lo);
            bA[0] = load_row_frag(br0 + k2, lo);
            bA[1] = load_row_frag(br1 + k2, lo);
        }

        #pragma unroll
        for (int i = 0; i < 4; ++i) acc[0][i] = wmma_bf16(aB[i], bB[0], acc[0][i]);
        #pragma unroll
        for (int i = 0; i < 4; ++i) acc[1][i] = wmma_bf16(aB[i], bB[1], acc[1][i]);
    }
}

// ---------------------------------------------------------------------------
// Kernel 1: QKV projection  qkv[m,n] = sum_k x[m,k] * w_qkv[n,k]
// Output scattered to per-head Q (pre-scaled) / K [B,H,N,64] and V^T [B,H,64,N].
// ---------------------------------------------------------------------------
__global__ __launch_bounds__(256) void qkv_gemm(
    const unsigned short* __restrict__ xb,    // [8192,768] bf16
    const unsigned short* __restrict__ wb,    // [2304,768] bf16 (row = out feat)
    unsigned short* __restrict__ qout,        // [48,2048,64]
    unsigned short* __restrict__ kout,        // [48,2048,64]
    unsigned short* __restrict__ vtout)       // [48,64,2048]
{
    const int lane = threadIdx.x & 31;
    const int wave = threadIdx.x >> 5;
    const int c  = lane & 15;
    const int hi = lane >> 4;
    const int lo = hi * 8;

    const int m0 = blockIdx.y * 64;                   // 128 m-tiles
    const int n0 = (blockIdx.x * 8 + wave) * 32;      // 72 n-jobs -> grid.x = 9

    v8f acc[2][4];
    #pragma unroll
    for (int j = 0; j < 2; ++j)
        #pragma unroll
        for (int i = 0; i < 4; ++i)
            acc[j][i] = (v8f){0.f,0.f,0.f,0.f,0.f,0.f,0.f,0.f};

    gemm_64x32_body(xb + (size_t)(m0 +  0 + c) * DIM,
                    xb + (size_t)(m0 + 16 + c) * DIM,
                    xb + (size_t)(m0 + 32 + c) * DIM,
                    xb + (size_t)(m0 + 48 + c) * DIM,
                    wb + (size_t)(n0 + c) * DIM,
                    wb + (size_t)(n0 + 16 + c) * DIM,
                    lo, acc);

    #pragma unroll
    for (int j = 0; j < 2; ++j) {
        const int n = n0 + 16 * j + c;     // segment uniform per 16-tile
        #pragma unroll
        for (int i = 0; i < 4; ++i) {
            #pragma unroll
            for (int r = 0; r < 8; ++r) {
                const int m    = m0 + 16 * i + r + 8 * hi;
                const int b    = m >> 11;
                const int npos = m & 2047;
                float val = acc[j][i][r];
                if (n < DIM) {                       // Q (pre-scaled)
                    int h = n >> 6, d = n & 63;
                    qout[(((size_t)(b * NHEAD + h) * SEQ) + npos) * HDIM + d] =
                        bf16_of(val * ATTN_SCALE);
                } else if (n < 2 * DIM) {            // K
                    int nn = n - DIM, h = nn >> 6, d = nn & 63;
                    kout[(((size_t)(b * NHEAD + h) * SEQ) + npos) * HDIM + d] =
                        bf16_of(val);
                } else {                             // V, stored transposed
                    int nn = n - 2 * DIM, h = nn >> 6, d = nn & 63;
                    vtout[(((size_t)(b * NHEAD + h) * HDIM) + d) * SEQ + npos] =
                        bf16_of(val);
                }
            }
        }
    }
}

// ---------------------------------------------------------------------------
// Kernel 2: flash attention. One wave per (b, h, 16-query tile).
// K tiles (32x64 bf16) streamed into LDS by the Tensor Data Mover, double
// buffered: next tile's TDM issued before s_wait_tensorcnt(1) on current.
// V fragments loaded from global before the softmax chain to overlap it.
// ---------------------------------------------------------------------------
__global__ __launch_bounds__(128) void attn_flash(
    const unsigned short* __restrict__ q,   // [48,2048,64]
    const unsigned short* __restrict__ k,   // [48,2048,64]
    const unsigned short* __restrict__ vt,  // [48,64,2048]
    unsigned short* __restrict__ attnb)     // [8192,768] = [B,N,H*64]
{
#if HAVE_TDM
    __shared__ __align__(16) unsigned short ldsK[4][2][32 * HDIM]; // 32 KB
#endif
    __shared__ __align__(16) unsigned short ldsP[4][16 * 32];      //  4 KB

    const int lane = threadIdx.x & 31;
    const int wave = threadIdx.x >> 5;
    const int c  = lane & 15;
    const int hi = lane >> 4;
    const int lo = hi * 8;

    const int job = blockIdx.x * 4 + wave;  // 6144 jobs
    const int qt  = job & 127;
    const int bh  = job >> 7;               // 0..47
    const int q0  = qt * 16;

    const unsigned short* qrow  = q  + ((size_t)bh * SEQ + q0 + c) * HDIM;
    const unsigned short* kbase = k  + (size_t)bh * SEQ * HDIM;
    const unsigned short* vbase = vt + (size_t)bh * HDIM * SEQ;
    unsigned short* myP = ldsP[wave];

    // Q fragments reused for all keys (A matrix, K-dim 64 = 2 x 32)
    const v16bf qa0 = load_row_frag(qrow, lo);
    const v16bf qa1 = load_row_frag(qrow + 32, lo);

    float rmax[8], rsum[8], alpha[8];
    v8f o[4];
    #pragma unroll
    for (int r = 0; r < 8; ++r) { rmax[r] = -3.0e38f; rsum[r] = 0.f; }
    #pragma unroll
    for (int dt = 0; dt < 4; ++dt) o[dt] = (v8f){0.f,0.f,0.f,0.f,0.f,0.f,0.f,0.f};

#if HAVE_TDM
    // prime buffer 0 with the first 32x64 K tile
    tdm_load_tile_2d(kbase, ldsK[wave][0], 32, HDIM, HDIM, SEQ);
#endif

    int ib = 0;
    for (int kt = 0; kt < SEQ; kt += 32, ib ^= 1) {
        // ---- acquire current K tile ------------------------------------
#if HAVE_TDM
        if (kt + 32 < SEQ) {
            tdm_load_tile_2d(kbase + (size_t)(kt + 32) * HDIM,
                             ldsK[wave][ib ^ 1], 32, HDIM, HDIM, SEQ);
            __builtin_amdgcn_s_wait_tensorcnt(1);   // current tile complete
        } else {
            __builtin_amdgcn_s_wait_tensorcnt(0);
        }
        asm volatile("" ::: "memory");              // keep LDS reads below wait
        const unsigned short* Kt = ldsK[wave][ib];
        v16bf kb00 = load_row_frag(Kt + (size_t)c * HDIM, lo);
        v16bf kb01 = load_row_frag(Kt + (size_t)c * HDIM + 32, lo);
        v16bf kb10 = load_row_frag(Kt + (size_t)(16 + c) * HDIM, lo);
        v16bf kb11 = load_row_frag(Kt + (size_t)(16 + c) * HDIM + 32, lo);
#else
        const unsigned short* k1 = kbase + (size_t)(kt + c) * HDIM;
        const unsigned short* k2 = kbase + (size_t)(kt + 16 + c) * HDIM;
        v16bf kb00 = load_row_frag(k1, lo);
        v16bf kb01 = load_row_frag(k1 + 32, lo);
        v16bf kb10 = load_row_frag(k2, lo);
        v16bf kb11 = load_row_frag(k2 + 32, lo);
        if (kt + 32 < SEQ)
            __builtin_prefetch(kbase + (size_t)(kt + 32 + lane) * HDIM, 0, 1);
#endif

        // ---- S = (Q*scale) . K^T ---------------------------------------
        v8f s1 = (v8f){0.f,0.f,0.f,0.f,0.f,0.f,0.f,0.f};
        s1 = wmma_bf16(qa0, kb00, s1);
        s1 = wmma_bf16(qa1, kb01, s1);
        v8f s2 = (v8f){0.f,0.f,0.f,0.f,0.f,0.f,0.f,0.f};
        s2 = wmma_bf16(qa0, kb10, s2);
        s2 = wmma_bf16(qa1, kb11, s2);

        // ---- V fragments issued early to overlap softmax ---------------
        v16bf vb[4];
        #pragma unroll
        for (int dt = 0; dt < 4; ++dt)
            vb[dt] = load_row_frag(vbase + (size_t)(dt * 16 + c) * SEQ + kt, lo);

        // ---- online softmax (rows r -> global row r + 8*hi) ------------
        #pragma unroll
        for (int r = 0; r < 8; ++r) {
            float t    = rowmax16(fmaxf(s1[r], s2[r]));
            float mnew = fmaxf(rmax[r], t);
            float a    = __expf(rmax[r] - mnew);
            float p1   = __expf(s1[r] - mnew);
            float p2   = __expf(s2[r] - mnew);
            rsum[r]  = a * rsum[r] + rowsum16(p1 + p2);
            rmax[r]  = mnew;
            alpha[r] = a;
            const int row = r + 8 * hi;
            myP[row * 32 + c]      = bf16_of(p1);
            myP[row * 32 + 16 + c] = bf16_of(p2);
        }

        // rescale running output
        #pragma unroll
        for (int dt = 0; dt < 4; ++dt)
            #pragma unroll
            for (int r = 0; r < 8; ++r) o[dt][r] *= alpha[r];

        // ---- O += P.V  (P via LDS: C-layout -> A-layout) ---------------
        v16bf pa = load_row_frag(myP + c * 32, lo);
        #pragma unroll
        for (int dt = 0; dt < 4; ++dt)
            o[dt] = wmma_bf16(pa, vb[dt], o[dt]);
    }

    // normalize + store to [B, N, H*64]
    const int b = bh / NHEAD, h = bh % NHEAD;
    #pragma unroll
    for (int dt = 0; dt < 4; ++dt) {
        #pragma unroll
        for (int r = 0; r < 8; ++r) {
            const int qi  = q0 + r + 8 * hi;
            const int col = h * HDIM + dt * 16 + c;
            attnb[((size_t)(b * SEQ + qi)) * DIM + col] = bf16_of(o[dt][r] / rsum[r]);
        }
    }
}

// ---------------------------------------------------------------------------
// Kernel 3: output projection  out[m,n] = sum_k attn[m,k]*w_proj[n,k] + b[n]
// ---------------------------------------------------------------------------
__global__ __launch_bounds__(256) void proj_gemm(
    const unsigned short* __restrict__ ab,   // [8192,768] bf16
    const unsigned short* __restrict__ wb,   // [768,768]  bf16
    const float* __restrict__ bias,          // [768]
    float* __restrict__ out)                 // [8192,768] f32
{
    const int lane = threadIdx.x & 31;
    const int wave = threadIdx.x >> 5;
    const int c  = lane & 15;
    const int hi = lane >> 4;
    const int lo = hi * 8;

    const int m0 = blockIdx.y * 64;
    const int n0 = (blockIdx.x * 8 + wave) * 32;   // 24 n-jobs -> grid.x = 3

    v8f acc[2][4];
    #pragma unroll
    for (int j = 0; j < 2; ++j)
        #pragma unroll
        for (int i = 0; i < 4; ++i)
            acc[j][i] = (v8f){0.f,0.f,0.f,0.f,0.f,0.f,0.f,0.f};

    gemm_64x32_body(ab + (size_t)(m0 +  0 + c) * DIM,
                    ab + (size_t)(m0 + 16 + c) * DIM,
                    ab + (size_t)(m0 + 32 + c) * DIM,
                    ab + (size_t)(m0 + 48 + c) * DIM,
                    wb + (size_t)(n0 + c) * DIM,
                    wb + (size_t)(n0 + 16 + c) * DIM,
                    lo, acc);

    #pragma unroll
    for (int j = 0; j < 2; ++j) {
        const int n = n0 + 16 * j + c;
        const float bv = bias[n];
        #pragma unroll
        for (int i = 0; i < 4; ++i) {
            #pragma unroll
            for (int r = 0; r < 8; ++r) {
                const int m = m0 + 16 * i + r + 8 * hi;
                out[(size_t)m * DIM + n] = acc[j][i][r] + bv;
            }
        }
    }
}

// ---------------------------------------------------------------------------
// Host-side launch
// ---------------------------------------------------------------------------
extern "C" void kernel_launch(void* const* d_in, const int* in_sizes, int n_in,
                              void* d_out, int out_size, void* d_ws, size_t ws_size,
                              hipStream_t stream) {
    const float* x      = (const float*)d_in[0];   // [4,2048,768]
    const float* w_qkv  = (const float*)d_in[1];   // [2304,768]
    const float* w_proj = (const float*)d_in[2];   // [768,768]
    const float* b_proj = (const float*)d_in[3];   // [768]
    float* out = (float*)d_out;

    // workspace layout (bf16 ushorts), all chunks 256B-aligned
    unsigned short* ws = (unsigned short*)d_ws;
    const size_t NX   = (size_t)MROWS * DIM;                 // 6,291,456
    const size_t NWQ  = (size_t)QKVDIM * DIM;                // 1,769,472
    const size_t NWP  = (size_t)DIM * DIM;                   //   589,824
    const size_t NHQK = (size_t)BATCH * NHEAD * SEQ * HDIM;  // 6,291,456
    unsigned short* xb     = ws;
    unsigned short* wqkvb  = xb + NX;
    unsigned short* wprojb = wqkvb + NWQ;
    unsigned short* qb     = wprojb + NWP;
    unsigned short* kb     = qb + NHQK;
    unsigned short* vtb    = kb + NHQK;
    unsigned short* attnb  = vtb + NHQK;

    // 0) convert inputs to bf16
    cvt_f32_bf16<<<(unsigned)((NX  + 255) / 256), 256, 0, stream>>>(x,      xb,     (int)NX);
    cvt_f32_bf16<<<(unsigned)((NWQ + 255) / 256), 256, 0, stream>>>(w_qkv,  wqkvb,  (int)NWQ);
    cvt_f32_bf16<<<(unsigned)((NWP + 255) / 256), 256, 0, stream>>>(w_proj, wprojb, (int)NWP);

    // 1) QKV projection: 72 n-jobs (9 blocks x 8 waves) x 128 m-tiles
    qkv_gemm<<<dim3(9, 128), 256, 0, stream>>>(xb, wqkvb, qb, kb, vtb);

    // 2) attention: 48 (b,h) x 128 q-tiles = 6144 wave-jobs / 4 waves per block
    attn_flash<<<dim3(1536), 128, 0, stream>>>(qb, kb, vtb, attnb);

    // 3) output projection: 24 n-jobs (3 blocks x 8 waves) x 128 m-tiles
    proj_gemm<<<dim3(3, 128), 256, 0, stream>>>(attnb, wprojb, b_proj, out);
}
